// net_5901285065253
// MI455X (gfx1250) — matrix operand order repaired and checked by reading, not compiled, not run on hardware
//
#include <hip/hip_runtime.h>
#include <hip/hip_bf16.h>
#include <math.h>

// ---- problem constants (from reference) ----
#define NNODE 50000
#define CIN   16
#define NT    4
#define K1    4
#define KEIG  128
#define NEDGE 1600000
#define HIDW  512
#define OUTW  256
#define KIN   256   // C*NT*K1

typedef __attribute__((ext_vector_type(16))) __bf16 v16bf;
typedef __attribute__((ext_vector_type(8)))  float  v8f;

// ------------------------------------------------------------------
// generic zero (float4 grid-stride)
// ------------------------------------------------------------------
__global__ void k_zero(float4* __restrict__ p, int n4) {
    int i = blockIdx.x * blockDim.x + threadIdx.x;
    int stride = gridDim.x * blockDim.x;
    float4 z = make_float4(0.f, 0.f, 0.f, 0.f);
    for (; i < n4; i += stride) p[i] = z;
}

// ------------------------------------------------------------------
// transpose + convert f32 [R][C] -> bf16 [C][R]
// ------------------------------------------------------------------
__global__ void k_transpose_bf16(__bf16* __restrict__ dst, const float* __restrict__ src,
                                 int R, int C) {
    int idx = blockIdx.x * blockDim.x + threadIdx.x;
    if (idx >= R * C) return;
    int r = idx / C, c = idx % C;
    dst[c * R + r] = (__bf16)src[idx];
}

// ------------------------------------------------------------------
// coeff[k][c] = sum_n evecs[n][k] * x[n][c]   (atomic split-N)
// thread t owns (k = t/2, c-range = (t&1)*8 .. +7)
// ------------------------------------------------------------------
__global__ __launch_bounds__(256) void k_coeff(float* __restrict__ coeff,
                                               const float* __restrict__ x,
                                               const float* __restrict__ evecs) {
    const int t = threadIdx.x;
    const int k = t >> 1;
    const int cbase = (t & 1) * 8;
    const int chunk = 391;                       // 128 * 391 >= 50000
    int n0 = blockIdx.x * chunk;
    int n1 = n0 + chunk; if (n1 > NNODE) n1 = NNODE;
    float acc[8];
#pragma unroll
    for (int j = 0; j < 8; ++j) acc[j] = 0.f;
    for (int n = n0; n < n1; ++n) {
        float ev = evecs[(size_t)n * KEIG + k];
        const float4* xp = reinterpret_cast<const float4*>(x + (size_t)n * CIN + cbase);
        float4 xa = xp[0], xb = xp[1];
        acc[0] += ev * xa.x; acc[1] += ev * xa.y; acc[2] += ev * xa.z; acc[3] += ev * xa.w;
        acc[4] += ev * xb.x; acc[5] += ev * xb.y; acc[6] += ev * xb.z; acc[7] += ev * xb.w;
    }
#pragma unroll
    for (int j = 0; j < 8; ++j) atomicAdd(&coeff[k * CIN + cbase + j], acc[j]);
}

// ------------------------------------------------------------------
// h0[n][t*16+c] = sum_k evecs[n][k] * exp(-evals[k]*taus[t]) * coeff[k][c]
// block handles 64 nodes; thread = (node, tscale)
// ------------------------------------------------------------------
__global__ __launch_bounds__(256) void k_diff(float* __restrict__ h0,
                                              const float* __restrict__ evecs,
                                              const float* __restrict__ evals,
                                              const float* __restrict__ taus,
                                              const float* __restrict__ coeff) {
    __shared__ float sc[NT * KEIG * CIN];        // 32 KB: sc[t][k][c]
    const int tid = threadIdx.x;
    for (int i = tid; i < NT * KEIG * CIN; i += 256) {
        int t = i >> 11;              // /2048
        int rem = i & 2047;
        int k = rem >> 4;
        int c = rem & 15;
        sc[i] = __expf(-evals[k] * taus[t]) * coeff[k * CIN + c];
    }
    __syncthreads();

    const int node = blockIdx.x * 64 + (tid >> 2);
    const int ts = tid & 3;
    if (node >= NNODE) return;

    float acc[16];
#pragma unroll
    for (int c = 0; c < 16; ++c) acc[c] = 0.f;
    const float* evrow = evecs + (size_t)node * KEIG;
    for (int k = 0; k < KEIG; ++k) {
        float ev = evrow[k];
        const float4* sp = reinterpret_cast<const float4*>(&sc[(ts * KEIG + k) * CIN]);
#pragma unroll
        for (int q = 0; q < 4; ++q) {
            float4 s4 = sp[q];
            acc[q * 4 + 0] += ev * s4.x;
            acc[q * 4 + 1] += ev * s4.y;
            acc[q * 4 + 2] += ev * s4.z;
            acc[q * 4 + 3] += ev * s4.w;
        }
    }
    float4* op = reinterpret_cast<float4*>(h0 + (size_t)node * (NT * CIN) + ts * CIN);
#pragma unroll
    for (int q = 0; q < 4; ++q)
        op[q] = make_float4(acc[q * 4], acc[q * 4 + 1], acc[q * 4 + 2], acc[q * 4 + 3]);
}

// ------------------------------------------------------------------
// edge scatter: h1[dst][k*64+c] += kvals[k][e] * h0[src][c]
// 64 threads per edge (c = tid%64), 4 edges per 256-thread block
// ------------------------------------------------------------------
__global__ __launch_bounds__(256) void k_edges(float* __restrict__ h1,
                                               const float* __restrict__ h0,
                                               const long long* __restrict__ eidx,
                                               const float* __restrict__ kvals) {
    const int tid = threadIdx.x;
    const int e = blockIdx.x * 4 + (tid >> 6);
    const int c = tid & 63;
    if (e >= NEDGE) return;
    const int src = (int)eidx[e];
    const int dst = (int)eidx[NEDGE + e];
    float v = h0[(size_t)src * 64 + c];
    float* base = h1 + (size_t)dst * KIN + c;
#pragma unroll
    for (int k = 0; k < K1; ++k)
        atomicAdd(base + k * 64, kvals[(size_t)k * NEDGE + e] * v);
}

// ------------------------------------------------------------------
// WMMA fragment loaders (ISA 7.12.2 layouts, wave32)
// ------------------------------------------------------------------
__device__ inline v16bf load_a_lds(const __bf16* base, int rowstride, int lane, int kb) {
    const int m = lane & 15;
    const int hi8 = (lane & 16) >> 1;            // 0 or 8
    const __bf16* row = base + m * rowstride + kb;
    union { unsigned int u[8]; v16bf v; } r;
#pragma unroll
    for (int g = 0; g < 8; ++g) {
        int k = ((g >> 2) << 4) + hi8 + ((g & 3) << 1);   // pairs {k,k+1}
        r.u[g] = *reinterpret_cast<const unsigned int*>(row + k);
    }
    return r.v;
}

__device__ inline v16bf load_b_gmem(const __bf16* __restrict__ wT, int Kdim,
                                    int lane, int ncol0, int kb) {
    const int n = ncol0 + (lane & 15);
    const int k0 = kb + ((lane & 16) >> 1);      // hi half-wave takes K+8 (and K+24)
    const uint4* p = reinterpret_cast<const uint4*>(wT + (size_t)n * Kdim + k0);
    union { uint4 q[2]; v16bf v; } u;
    u.q[0] = p[0];
    u.q[1] = p[2];
    return u.v;
}

// ------------------------------------------------------------------
// Fused MLP: h1[64,256] -> relu(.@W1+b1)[64,512] -> .@W2+b2 -> L2 norm
// 1 block = 64 nodes, 8 waves; each wave: 2 M-tiles x 4 N-tiles, so every
// B fragment feeds 2 WMMAs and W1T/W2T are streamed twice per block (two waves per N-group).
// ------------------------------------------------------------------
__global__ __launch_bounds__(256) void k_mlp(const float* __restrict__ h1,
                                             const __bf16* __restrict__ W1T,
                                             const float* __restrict__ b1,
                                             const __bf16* __restrict__ W2T,
                                             const float* __restrict__ b2,
                                             float* __restrict__ out) {
    // smem layout: [0,32K) A tile bf16 (dead after layer 1) ; ldsO f32 [0,64K)
    // aliases it ; ldsH bf16 at [64K,128K).
    __shared__ __align__(16) unsigned char smem[131072];
    __shared__ float ldsP[256];
    __shared__ float ldsInv[64];
    __bf16* ldsA = reinterpret_cast<__bf16*>(smem);            // [64][256]
    float*  ldsO = reinterpret_cast<float*>(smem);             // [64][256]
    __bf16* ldsH = reinterpret_cast<__bf16*>(smem + 65536);    // [64][512]

    const int tid = threadIdx.x;
    const int lane = tid & 31;
    const int wave = tid >> 5;
    const int node0 = blockIdx.x * 64;
    const int mpair = (wave & 1) * 2;        // first of this wave's 2 M-tiles
    const int ngrp  = wave >> 1;             // 0..3 : group of 4 N-tiles
    const int mhi   = (lane >> 4) * 8;
    const int ncl   = lane & 15;

    // stage input tile [64 x 256] f32 -> bf16 (zero-fill tail rows)
    for (int i = tid; i < 64 * KIN; i += 256) {
        int r = i >> 8;
        float v = (node0 + r < NNODE) ? h1[(size_t)node0 * KIN + i] : 0.f;
        ldsA[i] = (__bf16)v;
    }
    __syncthreads();

    // ---- layer 1: [64,256] x [256,512], two N-halves of 256 cols ----
    for (int half = 0; half < 2; ++half) {
        v8f acc[2][4] = {};
        for (int kb = 0; kb < KIN; kb += 32) {
            v16bf a0 = load_a_lds(ldsA + (mpair    ) * 16 * KIN, KIN, lane, kb);
            v16bf a1 = load_a_lds(ldsA + (mpair + 1) * 16 * KIN, KIN, lane, kb);
#pragma unroll
            for (int j = 0; j < 4; ++j) {
                int ncol0 = half * 256 + (ngrp * 4 + j) * 16;
                v16bf b = load_b_gmem(W1T, KIN, lane, ncol0, kb);
                acc[0][j] = __builtin_amdgcn_wmma_f32_16x16x32_bf16(
                    false, a0, false, b, (short)0, acc[0][j], false, false);
                acc[1][j] = __builtin_amdgcn_wmma_f32_16x16x32_bf16(
                    false, a1, false, b, (short)0, acc[1][j], false, false);
            }
        }
#pragma unroll
        for (int j = 0; j < 4; ++j) {
            int n = half * 256 + (ngrp * 4 + j) * 16 + ncl;
            float bias = b1[n];
#pragma unroll
            for (int mm = 0; mm < 2; ++mm) {
                int rbase = (mpair + mm) * 16 + mhi;
#pragma unroll
                for (int r = 0; r < 8; ++r) {
                    float v = acc[mm][j][r] + bias;
                    v = v > 0.f ? v : 0.f;               // ReLU
                    ldsH[(rbase + r) * HIDW + n] = (__bf16)v;
                }
            }
        }
    }
    __syncthreads();   // all A reads + H writes complete; ldsO may reuse A space

    // ---- layer 2: [64,512] x [512,256] ----
    {
        v8f acc[2][4] = {};
        for (int kb = 0; kb < HIDW; kb += 32) {
            v16bf a0 = load_a_lds(ldsH + (mpair    ) * 16 * HIDW, HIDW, lane, kb);
            v16bf a1 = load_a_lds(ldsH + (mpair + 1) * 16 * HIDW, HIDW, lane, kb);
#pragma unroll
            for (int j = 0; j < 4; ++j) {
                int ncol0 = (ngrp * 4 + j) * 16;
                v16bf b = load_b_gmem(W2T, HIDW, lane, ncol0, kb);
                acc[0][j] = __builtin_amdgcn_wmma_f32_16x16x32_bf16(
                    false, a0, false, b, (short)0, acc[0][j], false, false);
                acc[1][j] = __builtin_amdgcn_wmma_f32_16x16x32_bf16(
                    false, a1, false, b, (short)0, acc[1][j], false, false);
            }
        }
#pragma unroll
        for (int j = 0; j < 4; ++j) {
            int n = (ngrp * 4 + j) * 16 + ncl;
            float bias = b2[n];
#pragma unroll
            for (int mm = 0; mm < 2; ++mm) {
                int rbase = (mpair + mm) * 16 + mhi;
#pragma unroll
                for (int r = 0; r < 8; ++r)
                    ldsO[(rbase + r) * OUTW + n] = acc[mm][j][r] + bias;
            }
        }
    }
    __syncthreads();

    // ---- per-row L2 normalize: row = tid/4, 64-element segment each ----
    {
        int row = tid >> 2, seg = tid & 3;
        const float* rp = &ldsO[row * OUTW + seg * 64];
        float s = 0.f;
#pragma unroll
        for (int i = 0; i < 64; ++i) s += rp[i] * rp[i];
        ldsP[tid] = s;
    }
    __syncthreads();
    if (tid < 64) {
        float s = ldsP[tid * 4] + ldsP[tid * 4 + 1] + ldsP[tid * 4 + 2] + ldsP[tid * 4 + 3];
        float nrm = sqrtf(s);
        nrm = nrm > 1e-12f ? nrm : 1e-12f;
        ldsInv[tid] = 1.f / nrm;
    }
    __syncthreads();
    {
        int row = tid >> 2, seg = tid & 3;
        if (node0 + row < NNODE) {
            float inv = ldsInv[row];
            const float4* ip = reinterpret_cast<const float4*>(&ldsO[row * OUTW + seg * 64]);
            float4* op = reinterpret_cast<float4*>(out + (size_t)(node0 + row) * OUTW + seg * 64);
#pragma unroll
            for (int q = 0; q < 16; ++q) {
                float4 v = ip[q];
                op[q] = make_float4(v.x * inv, v.y * inv, v.z * inv, v.w * inv);
            }
        }
    }
}

// ------------------------------------------------------------------
// launch
// ------------------------------------------------------------------
extern "C" void kernel_launch(void* const* d_in, const int* in_sizes, int n_in,
                              void* d_out, int out_size, void* d_ws, size_t ws_size,
                              hipStream_t stream) {
    const float*     x     = (const float*)d_in[0];
    const float*     evals = (const float*)d_in[1];
    const float*     evecs = (const float*)d_in[2];
    const float*     taus  = (const float*)d_in[3];
    const float*     kvals = (const float*)d_in[4];
    const long long* eidx  = (const long long*)d_in[5];
    const float*     W1    = (const float*)d_in[6];
    const float*     b1    = (const float*)d_in[7];
    const float*     W2    = (const float*)d_in[8];
    const float*     b2    = (const float*)d_in[9];
    float* out = (float*)d_out;

    // workspace layout (bytes)
    char* ws = (char*)d_ws;
    float*  coeff = (float*)(ws);                               // 128*16 f32   = 8 KB
    float*  h0    = (float*)(ws + 8192);                        // N*64 f32     = 12.8 MB
    float*  h1    = (float*)(ws + 12808192);                    // N*256 f32    = 51.2 MB
    __bf16* W1T   = (__bf16*)(ws + 64008192);                   // 512x256 bf16 = 256 KB
    __bf16* W2T   = (__bf16*)(ws + 64270336);                   // 256x512 bf16 = 256 KB

    k_zero<<<8, 256, 0, stream>>>((float4*)coeff, (KEIG * CIN) / 4);
    k_zero<<<2048, 256, 0, stream>>>((float4*)h1, (NNODE * KIN) / 4);

    k_transpose_bf16<<<(KIN * HIDW) / 256, 256, 0, stream>>>(W1T, W1, KIN, HIDW);
    k_transpose_bf16<<<(HIDW * OUTW) / 256, 256, 0, stream>>>(W2T, W2, HIDW, OUTW);

    k_coeff<<<128, 256, 0, stream>>>(coeff, x, evecs);
    k_diff<<<(NNODE + 63) / 64, 256, 0, stream>>>(h0, evecs, evals, taus, coeff);
    k_edges<<<NEDGE / 4, 256, 0, stream>>>(h1, h0, eidx, kvals);
    k_mlp<<<(NNODE + 63) / 64, 256, 0, stream>>>(h1, W1T, b1, W2T, b2, out);
}
